// QuestMamba_30700426232246
// MI455X (gfx1250) — matrix operand order repaired and checked by reading, not compiled, not run on hardware
//
#include <hip/hip_runtime.h>

#define D_MODEL   1536
#define N_HEADS   48
#define HEAD_DIM  64
#define D_STATE   128
#define N_LAYER   4
#define K_CONV    4
#define D_INNER   3072
#define CONV_DIM  3328          // D_INNER + 2*D_STATE
#define D_IN_PROJ 6448          // 2*D_INNER + 2*D_STATE + N_HEADS
#define D_IN_PAD  6464          // padded to multiple of 64 for guard-free GEMM
#define CAT_DIM   16
#define BATCH     4
#define SEQ       512
#define NTOK      (BATCH*SEQ)   // 2048
#define EPS       1e-5f

typedef __attribute__((ext_vector_type(16))) __bf16 v16bf;
typedef __attribute__((ext_vector_type(8)))  __bf16 v8bf;
typedef __attribute__((ext_vector_type(8)))  float  v8f;
typedef unsigned short ushort_t;

__device__ __forceinline__ unsigned short f2bf(float f) {
  unsigned int u = __float_as_uint(f);
  u += 0x7FFFu + ((u >> 16) & 1u);          // round-to-nearest-even
  return (unsigned short)(u >> 16);
}
__device__ __forceinline__ float siluf(float v) { return v / (1.0f + __expf(-v)); }

// ---------------- embedding gather ----------------
__global__ void k_embed(const int* __restrict__ ids, const float* __restrict__ emb,
                        float* __restrict__ h) {
  int i = blockIdx.x * blockDim.x + threadIdx.x;
  if (i >= NTOK * D_MODEL) return;
  int tok = i / D_MODEL, d = i - tok * D_MODEL;
  h[i] = emb[(size_t)ids[tok] * D_MODEL + d];
}

// ---------------- f32 -> bf16 convert with zero row-padding to Npad ----------------
__global__ void k_f2bf_pad(const float* __restrict__ src, ushort_t* __restrict__ dst,
                           int N, int K, int Npad) {
  int i = blockIdx.x * blockDim.x + threadIdx.x;
  if (i >= Npad * K) return;
  int n = i / K;
  dst[i] = (n < N) ? f2bf(src[i]) : (ushort_t)0;
}

// ---------------- RMSNorm -> bf16 (per token, D_MODEL) ----------------
__global__ void __launch_bounds__(256)
k_rmsnorm_bf16(const float* __restrict__ x, const float* __restrict__ w,
               ushort_t* __restrict__ out) {
  __shared__ float red[256];
  int tok = blockIdx.x, t = threadIdx.x;
  const float* row = x + (size_t)tok * D_MODEL;
  float v[D_MODEL / 256];
  float ss = 0.f;
#pragma unroll
  for (int i = 0; i < D_MODEL / 256; ++i) { v[i] = row[t + i * 256]; ss += v[i] * v[i]; }
  red[t] = ss; __syncthreads();
  for (int s = 128; s > 0; s >>= 1) { if (t < s) red[t] += red[t + s]; __syncthreads(); }
  float scale = rsqrtf(red[0] / (float)D_MODEL + EPS);
  ushort_t* o = out + (size_t)tok * D_MODEL;
#pragma unroll
  for (int i = 0; i < D_MODEL / 256; ++i) { int d = t + i * 256; o[d] = f2bf(v[i] * scale * w[d]); }
}

// ---------------- RMSNorm -> f32 (final norm) ----------------
__global__ void __launch_bounds__(256)
k_rmsnorm_f32(const float* __restrict__ x, const float* __restrict__ w,
              float* __restrict__ out) {
  __shared__ float red[256];
  int tok = blockIdx.x, t = threadIdx.x;
  const float* row = x + (size_t)tok * D_MODEL;
  float v[D_MODEL / 256];
  float ss = 0.f;
#pragma unroll
  for (int i = 0; i < D_MODEL / 256; ++i) { v[i] = row[t + i * 256]; ss += v[i] * v[i]; }
  red[t] = ss; __syncthreads();
  for (int s = 128; s > 0; s >>= 1) { if (t < s) red[t] += red[t + s]; __syncthreads(); }
  float scale = rsqrtf(red[0] / (float)D_MODEL + EPS);
  float* o = out + (size_t)tok * D_MODEL;
#pragma unroll
  for (int i = 0; i < D_MODEL / 256; ++i) { int d = t + i * 256; o[d] = v[i] * scale * w[d]; }
}

// ------- bf16 WMMA GEMM:  C[M,Nc] = A[M,K] * W[Npad,K]^T (+res), rows >= Nc are zero-padding
// wave tile 32x64: 2 A-fragments x 4 B-fragments -> 8 WMMAs per k-step, guard-free inner loop.
// block = 8 waves -> 256x64 output per block.
__global__ void __launch_bounds__(256)
k_gemm_bf16(const __bf16* __restrict__ A, const __bf16* __restrict__ W,
            const float* __restrict__ res, float* __restrict__ C,
            int M, int Nc, int K) {
  const int wave = threadIdx.x >> 5;
  const int lane = threadIdx.x & 31;
  const int r    = lane & 15;
  const int hi   = lane >> 4;
  const int m0   = blockIdx.y * 256 + wave * 32;
  const int n0   = blockIdx.x * 64;

  v8f acc0[4] = {};
  v8f acc1[4] = {};
  const __bf16* Arow0 = A + (size_t)(m0 + r) * K;
  const __bf16* Arow1 = A + (size_t)(m0 + 16 + r) * K;
  const __bf16* Wrow  = W + (size_t)(n0 + r) * K;

  for (int k0 = 0; k0 < K; k0 += 32) {
    __builtin_prefetch(Arow0 + k0 + 256, 0, 3);   // global_prefetch_b8, near-cache
    __builtin_prefetch(Arow1 + k0 + 256, 0, 3);
    // A fragments: 16x32 (lane=r+16*hi holds row, K = 16hi..+15, matching B)
    v8bf a0lo = *(const v8bf*)(Arow0 + k0 + hi * 16);
    v8bf a0hi = *(const v8bf*)(Arow0 + k0 + 8 + hi * 16);
    v8bf a1lo = *(const v8bf*)(Arow1 + k0 + hi * 16);
    v8bf a1hi = *(const v8bf*)(Arow1 + k0 + 8 + hi * 16);
    v16bf af0, af1;
#pragma unroll
    for (int i = 0; i < 8; ++i) {
      af0[i] = a0lo[i]; af0[i + 8] = a0hi[i];
      af1[i] = a1lo[i]; af1[i + 8] = a1hi[i];
    }
#pragma unroll
    for (int t = 0; t < 4; ++t) {
      // B fragment: 32x16, lane=r+16*hi holds col n0+16t+r, K = k0+16hi..+15 (one 32B load)
      v16bf bf = *(const v16bf*)(Wrow + (size_t)t * 16 * K + k0 + hi * 16);
      acc0[t] = __builtin_amdgcn_wmma_f32_16x16x32_bf16(
          false, af0, false, bf, (short)0, acc0[t], false, false);
      acc1[t] = __builtin_amdgcn_wmma_f32_16x16x32_bf16(
          false, af1, false, bf, (short)0, acc1[t], false, false);
    }
  }
#pragma unroll
  for (int t = 0; t < 4; ++t) {
    int n = n0 + t * 16 + r;
    if (n >= Nc) continue;
#pragma unroll
    for (int j = 0; j < 8; ++j) {
      int m = m0 + hi * 8 + j;
      size_t idx0 = (size_t)m * Nc + n;
      size_t idx1 = (size_t)(m + 16) * Nc + n;
      float v0 = acc0[t][j];
      float v1 = acc1[t][j];
      if (res) { v0 += res[idx0]; v1 += res[idx1]; }
      C[idx0] = v0;
      C[idx1] = v1;
    }
  }
}

// ---------------- depthwise causal conv(k=4) + bias + SiLU ----------------
__global__ void k_conv(const float* __restrict__ zx, const float* __restrict__ cw,
                       const float* __restrict__ cb, float* __restrict__ out) {
  int i = blockIdx.x * blockDim.x + threadIdx.x;
  if (i >= NTOK * CONV_DIM) return;
  int c = i % CONV_DIM; int tok = i / CONV_DIM;
  int l = tok % SEQ, b = tok / SEQ;
  float acc = cb[c];
#pragma unroll
  for (int j = 0; j < K_CONV; ++j) {
    int ls = l - (K_CONV - 1) + j;
    if (ls >= 0)
      acc += cw[c * K_CONV + j] * zx[((size_t)(b * SEQ + ls)) * D_IN_PROJ + D_INNER + c];
  }
  out[i] = siluf(acc);
}

// ---------------- dt = softplus(raw + bias) ----------------
__global__ void k_dt(const float* __restrict__ zx, const float* __restrict__ dtb,
                     float* __restrict__ out) {
  int i = blockIdx.x * blockDim.x + threadIdx.x;
  if (i >= NTOK * N_HEADS) return;
  int hh = i % N_HEADS; int tok = i / N_HEADS;
  float v = zx[(size_t)tok * D_IN_PROJ + (D_INNER + CONV_DIM) + hh] + dtb[hh];
  out[i] = (v > 20.f) ? v : log1pf(__expf(v));
}

// ---------------- SSD sequential scan: one block per (batch, head) ----------------
__global__ void __launch_bounds__(256)
k_scan(const float* __restrict__ xbc, const float* __restrict__ dt,
       const float* __restrict__ A_log, const float* __restrict__ Dv,
       float* __restrict__ y) {
  __shared__ float sB[D_STATE], sC[D_STATE];
  int b = blockIdx.x / N_HEADS, hh = blockIdx.x % N_HEADS;
  int t = threadIdx.x;
  int p = t >> 2, nq = t & 3;
  float A  = -__expf(A_log[hh]);
  float Dh = Dv[hh];
  float s[32];
#pragma unroll
  for (int i = 0; i < 32; ++i) s[i] = 0.f;

  for (int l = 0; l < SEQ; ++l) {
    size_t rowb = (size_t)(b * SEQ + l);
    const float* xr = xbc + rowb * CONV_DIM;
    if (t < D_STATE)          sB[t] = xr[D_INNER + t];
    else if (t < 2 * D_STATE) sC[t - D_STATE] = xr[D_INNER + D_STATE + (t - D_STATE)];
    __syncthreads();

    float dts = dt[rowb * N_HEADS + hh];
    float dec = __expf(dts * A);
    float xv  = xr[hh * HEAD_DIM + p];
    float dtx = dts * xv;
    float part = 0.f;
    const float* Bp = sB + nq * 32;
    const float* Cp = sC + nq * 32;
#pragma unroll
    for (int i = 0; i < 32; ++i) {
      s[i] = s[i] * dec + dtx * Bp[i];
      part += s[i] * Cp[i];
    }
    part += __shfl_xor(part, 1, 32);
    part += __shfl_xor(part, 2, 32);
    if (nq == 0) y[rowb * D_INNER + hh * HEAD_DIM + p] = part + Dh * xv;
    __syncthreads();
  }
}

// ---------------- gated RMSNorm -> bf16: rmsnorm(y * silu(z)) * gw ----------------
__global__ void __launch_bounds__(256)
k_gatednorm(const float* __restrict__ zx, const float* __restrict__ y,
            const float* __restrict__ gw, ushort_t* __restrict__ out) {
  __shared__ float red[256];
  int tok = blockIdx.x, t = threadIdx.x;
  const float* zr = zx + (size_t)tok * D_IN_PROJ;   // z = first D_INNER of zxbcdt
  const float* yr = y + (size_t)tok * D_INNER;
  float g[D_INNER / 256];
  float ss = 0.f;
#pragma unroll
  for (int i = 0; i < D_INNER / 256; ++i) {
    int d = t + i * 256;
    float gv = yr[d] * siluf(zr[d]);
    g[i] = gv; ss += gv * gv;
  }
  red[t] = ss; __syncthreads();
  for (int s2 = 128; s2 > 0; s2 >>= 1) { if (t < s2) red[t] += red[t + s2]; __syncthreads(); }
  float scale = rsqrtf(red[0] / (float)D_INNER + EPS);
  ushort_t* o = out + (size_t)tok * D_INNER;
#pragma unroll
  for (int i = 0; i < D_INNER / 256; ++i) { int d = t + i * 256; o[d] = f2bf(g[i] * scale * gw[d]); }
}

// ---------------- masked mean pool over sequence ----------------
__global__ void __launch_bounds__(256)
k_pool(const float* __restrict__ hn, const int* __restrict__ mask,
       float* __restrict__ text) {
  int b = blockIdx.x;
  int d = blockIdx.y * 256 + threadIdx.x;
  float acc = 0.f, cnt = 0.f;
  for (int l = 0; l < SEQ; ++l) {
    float m = (float)mask[b * SEQ + l];
    cnt += m;
    acc += m * hn[((size_t)(b * SEQ + l)) * D_MODEL + d];
  }
  text[b * D_MODEL + d] = acc / fmaxf(cnt, 1e-9f);
}

// ---------------- heads: one wave per (batch, output) ----------------
__global__ void k_head(const float* __restrict__ text, const float* __restrict__ cat_emb,
                       const int* __restrict__ cats,
                       const float* __restrict__ qw, const float* __restrict__ qb,
                       const float* __restrict__ aw, const float* __restrict__ ab,
                       float* __restrict__ out) {
  int b = blockIdx.x / 30, o = blockIdx.x % 30;
  int lane = threadIdx.x;
  const float* w; float bias;
  if (o < 21) { w = qw + (size_t)o * (D_MODEL + CAT_DIM); bias = qb[o]; }
  else        { w = aw + (size_t)(o - 21) * (D_MODEL + CAT_DIM); bias = ab[o - 21]; }
  const float* ce = cat_emb + (size_t)cats[b] * CAT_DIM;
  float acc = 0.f;
  for (int i = lane; i < D_MODEL + CAT_DIM; i += 32) {
    float f = (i < D_MODEL) ? text[b * D_MODEL + i] : ce[i - D_MODEL];
    acc += f * w[i];
  }
#pragma unroll
  for (int m = 16; m >= 1; m >>= 1) acc += __shfl_xor(acc, m, 32);
  if (lane == 0) out[b * 30 + o] = acc + bias;
}

// ---------------- host launch ----------------
extern "C" void kernel_launch(void* const* d_in, const int* in_sizes, int n_in,
                              void* d_out, int out_size, void* d_ws, size_t ws_size,
                              hipStream_t stream) {
  const int*   ids      = (const int*)d_in[0];
  const int*   mask     = (const int*)d_in[1];
  const int*   cats     = (const int*)d_in[2];
  const float* emb      = (const float*)d_in[3];
  const float* norm_w   = (const float*)d_in[4];
  const float* in_proj  = (const float*)d_in[5];
  const float* conv_w   = (const float*)d_in[6];
  const float* conv_b   = (const float*)d_in[7];
  const float* dt_bias  = (const float*)d_in[8];
  const float* A_log    = (const float*)d_in[9];
  const float* Dv       = (const float*)d_in[10];
  const float* gnorm_w  = (const float*)d_in[11];
  const float* out_proj = (const float*)d_in[12];
  const float* final_w  = (const float*)d_in[13];
  const float* cat_emb  = (const float*)d_in[14];
  const float* qw       = (const float*)d_in[15];
  const float* qb       = (const float*)d_in[16];
  const float* aw       = (const float*)d_in[17];
  const float* ab       = (const float*)d_in[18];

  char* ws = (char*)d_ws;
  float*    h    = (float*)   (ws + 0);            // 2048*1536*4  = 12,582,912
  ushort_t* xn   = (ushort_t*)(ws + 12582912);     // 2048*1536*2  =  6,291,456
  float*    zx   = (float*)   (ws + 18874368);     // 2048*6448*4  = 52,822,016
  float*    xbc  = (float*)   (ws + 71696384);     // 2048*3328*4  = 27,262,976
  float*    dtb  = (float*)   (ws + 98959360);     // 2048*48*4    =    393,216
  float*    ybuf = (float*)   (ws + 99352576);     // 2048*3072*4  = 25,165,824 (reused as hn)
  ushort_t* ybf  = (ushort_t*)(ws + 124518400);    // 2048*3072*2  = 12,582,912
  ushort_t* wbf  = (ushort_t*)(ws + 137101312);    // 6464*1536*2  = 19,857,408 (per-layer reuse)
  float*    text = (float*)   (ws + 156958720);    // 4*1536*4

  // embedding
  k_embed<<<(NTOK * D_MODEL + 255) / 256, 256, 0, stream>>>(ids, emb, h);

  for (int layer = 0; layer < N_LAYER; ++layer) {
    const float* w_in  = in_proj  + (size_t)layer * D_IN_PROJ * D_MODEL;
    const float* w_out = out_proj + (size_t)layer * D_MODEL * D_INNER;

    // in_proj weights -> bf16, zero-padded to D_IN_PAD rows
    k_f2bf_pad<<<(D_IN_PAD * D_MODEL + 255) / 256, 256, 0, stream>>>(
        w_in, wbf, D_IN_PROJ, D_MODEL, D_IN_PAD);
    // rmsnorm(h) -> bf16
    k_rmsnorm_bf16<<<NTOK, 256, 0, stream>>>(h, norm_w + layer * D_MODEL, xn);
    // zxbcdt = xn @ w_in^T
    {
      dim3 g(D_IN_PAD / 64, NTOK / 256);
      k_gemm_bf16<<<g, 256, 0, stream>>>((const __bf16*)xn, (const __bf16*)wbf,
                                         nullptr, zx, NTOK, D_IN_PROJ, D_MODEL);
    }
    // causal depthwise conv + silu
    k_conv<<<(NTOK * CONV_DIM + 255) / 256, 256, 0, stream>>>(
        zx, conv_w + (size_t)layer * CONV_DIM * K_CONV, conv_b + (size_t)layer * CONV_DIM, xbc);
    // dt = softplus
    k_dt<<<(NTOK * N_HEADS + 255) / 256, 256, 0, stream>>>(zx, dt_bias + layer * N_HEADS, dtb);
    // SSD scan
    k_scan<<<BATCH * N_HEADS, 256, 0, stream>>>(
        xbc, dtb, A_log + layer * N_HEADS, Dv + layer * N_HEADS, ybuf);
    // gated rmsnorm -> bf16
    k_gatednorm<<<NTOK, 256, 0, stream>>>(zx, ybuf, gnorm_w + (size_t)layer * D_INNER, ybf);
    // out_proj weights -> bf16 (N=1536 already multiple of 64)
    k_f2bf_pad<<<(D_MODEL * D_INNER + 255) / 256, 256, 0, stream>>>(
        w_out, wbf, D_MODEL, D_INNER, D_MODEL);
    // h += ybf @ w_out^T   (residual fused, in-place)
    {
      dim3 g(D_MODEL / 64, NTOK / 256);
      k_gemm_bf16<<<g, 256, 0, stream>>>((const __bf16*)ybf, (const __bf16*)wbf,
                                         h, h, NTOK, D_MODEL, D_INNER);
    }
  }

  // final norm (into reused ybuf region), masked pool, heads
  float* hn = ybuf;
  k_rmsnorm_f32<<<NTOK, 256, 0, stream>>>(h, final_w, hn);
  {
    dim3 g(BATCH, D_MODEL / 256);
    k_pool<<<g, 256, 0, stream>>>(hn, mask, text);
  }
  k_head<<<BATCH * 30, 32, 0, stream>>>(text, cat_emb, cats, qw, qb, aw, ab, (float*)d_out);
}
